// Model_79285096284611
// MI455X (gfx1250) — matrix operand order repaired and checked by reading, not compiled, not run on hardware
//
#include <hip/hip_runtime.h>
#include <hip/hip_bf16.h>
#include <cstdint>

// ---------------------------------------------------------------------------
// CDNA5 (gfx1250) fused implementation.
// All sequences in the reference have length 1, so every conv is a center-tap
// matmul and every MHA is just its output projection. The whole model is a
// per-row MLP chain batched over B=16384 rows -> tiled 16 rows per wave and
// executed with v_wmma_f32_16x16x32_bf16.
//
// Everything constant (bf16-packed weights, folded BN epilogue pairs, biases)
// lives in the workspace at compile-time-constant offsets, so the fused kernel
// carries exactly two pointers: all loads become SADDR(ws)+imm24 addressing
// and SGPR pressure stays minimal (no readlane spills).
// ---------------------------------------------------------------------------

typedef __attribute__((ext_vector_type(16))) __bf16 v16bf;
typedef __attribute__((ext_vector_type(8)))  float  v8f;

union BF16x16 { uint4 u[2]; v16bf v; };

// ---------------- workspace layout (compile-time constants) ----------------
constexpr size_t AL256(size_t x) { return (x + 255) & ~(size_t)255; }
// bf16 weights (K-major, row = K, 16-col tiles contiguous)
constexpr size_t OFF_WG    = 0;                                    // 832 x 768
constexpr size_t OFF_W1RC1 = OFF_WG    + AL256(832 * 768 * 2);     // 896 x 128
constexpr size_t OFF_W1C1  = OFF_W1RC1 + AL256(896 * 128 * 2);
constexpr size_t OFF_W1RC2 = OFF_W1C1  + AL256(896 * 128 * 2);     // 128 x 128
constexpr size_t OFF_W1C2  = OFF_W1RC2 + AL256(128 * 128 * 2);
constexpr size_t OFF_W2RC1 = OFF_W1C2  + AL256(128 * 128 * 2);     // 128 x 256
constexpr size_t OFF_W2C1  = OFF_W2RC1 + AL256(128 * 256 * 2);
constexpr size_t OFF_W2RC2 = OFF_W2C1  + AL256(128 * 256 * 2);     // 256 x 256
constexpr size_t OFF_W2C2  = OFF_W2RC2 + AL256(256 * 256 * 2);
constexpr size_t OFF_W3RC1 = OFF_W2C2  + AL256(256 * 256 * 2);
constexpr size_t OFF_W3C1  = OFF_W3RC1 + AL256(256 * 256 * 2);
constexpr size_t OFF_W3RC2 = OFF_W3C1  + AL256(256 * 256 * 2);
constexpr size_t OFF_W3C2  = OFF_W3RC2 + AL256(256 * 256 * 2);
constexpr size_t OFF_WATT  = OFF_W3C2  + AL256(256 * 256 * 2);     // 256 x 256
constexpr size_t OFF_WATT1 = OFF_WATT  + AL256(256 * 256 * 2);     // 512 x 512
constexpr size_t OFF_WFC1  = OFF_WATT1 + AL256(512 * 512 * 2);     // 512 x 64
constexpr size_t OFF_WFC2  = OFF_WFC1  + AL256(512 * 64 * 2);      // 64 x 16
constexpr size_t OFF_WFC3  = OFF_WFC2  + AL256(64 * 16 * 2);       // 32 x 16
constexpr size_t OFF_WEND  = OFF_WFC3  + AL256(32 * 16 * 2);
// fp32 constants
constexpr size_t OFF_BG    = OFF_WEND;                             // [768] bih concat
constexpr size_t OFF_BHH   = OFF_BG    + AL256(768 * 4);           // [768] bhh concat
constexpr size_t OFF_EP1A  = OFF_BHH   + AL256(768 * 4);           // rb1 float2[128] x3
constexpr size_t OFF_EP2A  = OFF_EP1A  + AL256(128 * 8);
constexpr size_t OFF_EP3A  = OFF_EP2A  + AL256(128 * 8);
constexpr size_t OFF_EP1B  = OFF_EP3A  + AL256(128 * 8);           // rb2 float2[256] x3
constexpr size_t OFF_EP2B  = OFF_EP1B  + AL256(256 * 8);
constexpr size_t OFF_EP3B  = OFF_EP2B  + AL256(256 * 8);
constexpr size_t OFF_EP1C  = OFF_EP3B  + AL256(256 * 8);           // rb3 float2[256] x3
constexpr size_t OFF_EP2C  = OFF_EP1C  + AL256(256 * 8);
constexpr size_t OFF_EP3C  = OFF_EP2C  + AL256(256 * 8);
constexpr size_t OFF_B1RC1 = OFF_EP3C  + AL256(256 * 8);           // raw residual biases
constexpr size_t OFF_B1RC2 = OFF_B1RC1 + AL256(128 * 4);
constexpr size_t OFF_B2RC1 = OFF_B1RC2 + AL256(128 * 4);
constexpr size_t OFF_B2RC2 = OFF_B2RC1 + AL256(256 * 4);
constexpr size_t OFF_B3RC1 = OFF_B2RC2 + AL256(256 * 4);
constexpr size_t OFF_B3RC2 = OFF_B3RC1 + AL256(256 * 4);
constexpr size_t OFF_ATTB  = OFF_B3RC2 + AL256(256 * 4);
constexpr size_t OFF_ATT1B = OFF_ATTB  + AL256(256 * 4);
constexpr size_t OFF_FC1B  = OFF_ATT1B + AL256(512 * 4);
constexpr size_t OFF_FC2B  = OFF_FC1B  + AL256(64 * 4);
constexpr size_t OFF_FC3B  = OFF_FC2B  + AL256(16 * 4);
constexpr size_t OFF_XB    = OFF_FC3B  + AL256(16 * 4);            // [B][1728] bf16

__device__ __forceinline__ float lrelu_f(float x, float s) { return x >= 0.f ? x : x * s; }
__device__ __forceinline__ float sigmoid_f(float x) { return 1.f / (1.f + __expf(-x)); }

// One 16xK (A, row-major bf16, global or LDS) x Kx(NT*16) (Wt, row-major KxNpad
// bf16) tile-chunk. A fragment: lane<16 -> M=lane, K0..7 & K16..23 (two b128 at
// +0/+32B); lane>=16 -> K8..15 & K24..31. B fragment: lane&15 = N column,
// elements = same K order as A (strided scalar loads).
template<int NT>
__device__ __forceinline__ void mm_block(const __bf16* A, int aStride,
                                         const __bf16* Wt, int Npad,
                                         int Ktiles, int nt0, v8f* acc)
{
  const int lane = threadIdx.x;
  const int row  = lane & 15;
  const int half = lane >> 4;
  const v8f vzero = {0.f, 0.f, 0.f, 0.f, 0.f, 0.f, 0.f, 0.f};
#pragma unroll
  for (int i = 0; i < NT; ++i) acc[i] = vzero;

  for (int kt = 0; kt < Ktiles; ++kt) {
    BF16x16 a;
    const __bf16* pa = A + (size_t)row * aStride + kt * 32 + half * 8;
    a.u[0] = *(const uint4*)(pa);
    a.u[1] = *(const uint4*)(pa + 16);
    const __bf16* pb = Wt + (size_t)(kt * 32 + lane) * Npad + nt0 * 16;
    if (kt + 1 < Ktiles)  // stream next weight K-tile through L2
      __builtin_prefetch((const void*)(pb + (size_t)32 * Npad), 0, 0);
#pragma unroll
    for (int i = 0; i < NT; ++i) {
      BF16x16 b;
      const __bf16* pbi = Wt + (size_t)(kt * 32 + half * 8) * Npad + (nt0 + i) * 16 + row;
#pragma unroll
      for (int s = 0; s < 8; ++s) {
        b.v[s] = pbi[(size_t)s * Npad];
        b.v[8 + s] = pbi[(size_t)(16 + s) * Npad];
      }
      acc[i] = __builtin_amdgcn_wmma_f32_16x16x32_bf16(
          false, a.v, false, b.v, (short)0, acc[i], false, false);
    }
  }
}

struct RBO {          // per-res-block constant ws offsets (folded at inline time)
  size_t wrc1, wc1, wrc2, wc2, brc1, brc2, ep1, ep2, ep3;
};

// Length-1 res_block: 6 center-tap matmuls + folded BN + residual + lrelu(.02)
// bx/by: ping-pong LDS activation regions (row stride 768 bf16), sR: fp32
// residual (row stride 256). Single wave -> in-order LDS, no barriers needed.
__device__ __forceinline__ void res_block(const char* ws,
                                          const __bf16* inB, int inStride, int Ktin,
                                          __bf16* bx, __bf16* by, float* sR,
                                          int Cout, RBO o)
{
  const int lane  = threadIdx.x;
  const int nl    = lane & 15;
  const int mbase = (lane >> 4) * 8;
  const int NC    = Cout >> 7;   // 128-col chunks
  const int Kto   = Cout >> 5;   // K tiles when the input is 'out'
  const __bf16* Wrc1 = (const __bf16*)(ws + o.wrc1);
  const __bf16* Wc1  = (const __bf16*)(ws + o.wc1);
  const __bf16* Wrc2 = (const __bf16*)(ws + o.wrc2);
  const __bf16* Wc2  = (const __bf16*)(ws + o.wc2);
  const float*  brc1 = (const float*)(ws + o.brc1);
  const float*  brc2 = (const float*)(ws + o.brc2);
  const float2* ep1  = (const float2*)(ws + o.ep1);
  const float2* ep2  = (const float2*)(ws + o.ep2);
  const float2* ep3  = (const float2*)(ws + o.ep3);
  v8f acc[8];

  // stage 1:  R = rc1(in);  o1 = lrelu(ep1*acc+shift + R) -> bx
  for (int c = 0; c < NC; ++c) {
    mm_block<8>(inB, inStride, Wrc1, Cout, Ktin, c * 8, acc);
#pragma unroll
    for (int i = 0; i < 8; ++i) { int n = (c * 8 + i) * 16 + nl; float bb = brc1[n];
#pragma unroll
      for (int r = 0; r < 8; ++r) sR[(mbase + r) * 256 + n] = acc[i][r] + bb; }
    mm_block<8>(inB, inStride, Wc1, Cout, Ktin, c * 8, acc);
#pragma unroll
    for (int i = 0; i < 8; ++i) { int n = (c * 8 + i) * 16 + nl; float2 e = ep1[n];
#pragma unroll
      for (int r = 0; r < 8; ++r) {
        float v = e.x * acc[i][r] + e.y + sR[(mbase + r) * 256 + n];
        bx[(mbase + r) * 768 + n] = (__bf16)lrelu_f(v, 0.02f); } }
  }
  // stage 2:  R += rc2(o1);  o2 = lrelu(ep2*acc+shift + R) -> by
  for (int c = 0; c < NC; ++c) {
    mm_block<8>(bx, 768, Wrc2, Cout, Kto, c * 8, acc);
#pragma unroll
    for (int i = 0; i < 8; ++i) { int n = (c * 8 + i) * 16 + nl; float bb = brc2[n];
#pragma unroll
      for (int r = 0; r < 8; ++r) sR[(mbase + r) * 256 + n] += acc[i][r] + bb; }
    mm_block<8>(bx, 768, Wc2, Cout, Kto, c * 8, acc);
#pragma unroll
    for (int i = 0; i < 8; ++i) { int n = (c * 8 + i) * 16 + nl; float2 e = ep2[n];
#pragma unroll
      for (int r = 0; r < 8; ++r) {
        float v = e.x * acc[i][r] + e.y + sR[(mbase + r) * 256 + n];
        by[(mbase + r) * 768 + n] = (__bf16)lrelu_f(v, 0.02f); } }
  }
  // stage 3:  R += rc2(o2);  o3 = lrelu(ep3*acc+shift + R) -> bx
  for (int c = 0; c < NC; ++c) {
    mm_block<8>(by, 768, Wrc2, Cout, Kto, c * 8, acc);
#pragma unroll
    for (int i = 0; i < 8; ++i) { int n = (c * 8 + i) * 16 + nl; float bb = brc2[n];
#pragma unroll
      for (int r = 0; r < 8; ++r) sR[(mbase + r) * 256 + n] += acc[i][r] + bb; }
    mm_block<8>(by, 768, Wc2, Cout, Kto, c * 8, acc);
#pragma unroll
    for (int i = 0; i < 8; ++i) { int n = (c * 8 + i) * 16 + nl; float2 e = ep3[n];
#pragma unroll
      for (int r = 0; r < 8; ++r) {
        float v = e.x * acc[i][r] + e.y + sR[(mbase + r) * 256 + n];
        bx[(mbase + r) * 768 + n] = (__bf16)lrelu_f(v, 0.02f); } }
  }
}

__global__ __launch_bounds__(32) void fused_model(const char* __restrict__ ws,
                                                  float* __restrict__ out)
{
  __shared__ __attribute__((aligned(16))) __bf16 sG[16 * 768]; // gi / P0(0..255) P1(256..511) / z'
  __shared__ __attribute__((aligned(16))) __bf16 sZ[16 * 512]; // h|a1 (0..255), a2 (256..511)
  __shared__ __attribute__((aligned(16))) float  sR[16 * 256]; // residual fp32

  const int lane  = threadIdx.x;
  const int nl    = lane & 15;
  const int mbase = (lane >> 4) * 8;
  const int r0    = blockIdx.x * 16;
  const __bf16* xb = (const __bf16*)(ws + OFF_XB);
  v8f acc[8];

  // ---- GRU gate pre-activations: gi = bn(x1) @ [fw_wih;bw_wih]^T + bih -> sG
  {
    const __bf16* Ax1 = xb + (size_t)r0 * 1728;
    const __bf16* Wg  = (const __bf16*)(ws + OFF_WG);
    const float*  bg  = (const float*)(ws + OFF_BG);
    for (int c = 0; c < 6; ++c) {
      mm_block<8>(Ax1, 1728, Wg, 768, 26, c * 8, acc);
#pragma unroll
      for (int i = 0; i < 8; ++i) { int n = (c * 8 + i) * 16 + nl; float bb = bg[n];
#pragma unroll
        for (int r = 0; r < 8; ++r) sG[(mbase + r) * 768 + n] = (__bf16)(acc[i][r] + bb); }
    }
  }
  // ---- GRU gates (h_prev = 0): h = (1-z)*tanh(gn + r*bhh_n) -> sZ[:, :256]
  {
    const float* bhh = (const float*)(ws + OFF_BHH);
    for (int t = lane; t < 16 * 256; t += 32) {
      int row = t >> 8, j = t & 255;
      int dir = j >> 7, jj = j & 127;
      int base = dir * 384;
      float g0 = (float)sG[row * 768 + base + jj];
      float g1 = (float)sG[row * 768 + base + 128 + jj];
      float g2 = (float)sG[row * 768 + base + 256 + jj];
      float rg = sigmoid_f(g0 + bhh[base + jj]);
      float zg = sigmoid_f(g1 + bhh[base + 128 + jj]);
      float ng = tanhf(g2 + rg * bhh[base + 256 + jj]);
      sZ[row * 512 + j] = (__bf16)((1.f - zg) * ng);
    }
  }
  // ---- conv trunk on bn(x2) (length-1 signal => center-tap matmuls)
  constexpr RBO rb1 = { OFF_W1RC1, OFF_W1C1, OFF_W1RC2, OFF_W1C2,
                        OFF_B1RC1, OFF_B1RC2, OFF_EP1A, OFF_EP2A, OFF_EP3A };
  constexpr RBO rb2 = { OFF_W2RC1, OFF_W2C1, OFF_W2RC2, OFF_W2C2,
                        OFF_B2RC1, OFF_B2RC2, OFF_EP1B, OFF_EP2B, OFF_EP3B };
  constexpr RBO rb3 = { OFF_W3RC1, OFF_W3C1, OFF_W3RC2, OFF_W3C2,
                        OFF_B3RC1, OFF_B3RC2, OFF_EP1C, OFF_EP2C, OFF_EP3C };
  res_block(ws, xb + (size_t)r0 * 1728 + 832, 1728, 28, sG,       sG + 256, sR, 128, rb1);
  res_block(ws, sG,       768, 4,                sG + 256, sG,       sR, 256, rb2);
  res_block(ws, sG + 256, 768, 8,                sG,       sG + 256, sR, 256, rb3);
  // ---- a2 = att(x2_out) -> sZ[:, 256:512]  (L=1 -> MHA == out-projection)
  {
    const __bf16* Watt = (const __bf16*)(ws + OFF_WATT);
    const float* attb  = (const float*)(ws + OFF_ATTB);
    for (int c = 0; c < 2; ++c) {
      mm_block<8>(sG, 768, Watt, 256, 8, c * 8, acc);
#pragma unroll
      for (int i = 0; i < 8; ++i) { int n = (c * 8 + i) * 16 + nl; float bb = attb[n];
#pragma unroll
        for (int r = 0; r < 8; ++r) sZ[(mbase + r) * 512 + 256 + n] = (__bf16)(acc[i][r] + bb); }
    }
    // ---- a1 = att(h) -> sG[:, 256:512] (P1 dead)
    for (int c = 0; c < 2; ++c) {
      mm_block<8>(sZ, 512, Watt, 256, 8, c * 8, acc);
#pragma unroll
      for (int i = 0; i < 8; ++i) { int n = (c * 8 + i) * 16 + nl; float bb = attb[n];
#pragma unroll
        for (int r = 0; r < 8; ++r) sG[(mbase + r) * 768 + 256 + n] = (__bf16)(acc[i][r] + bb); }
    }
  }
  // move a1 over dead h: z = [a1, a2] contiguous in sZ
  for (int t = lane; t < 16 * 256; t += 32)
    sZ[(t >> 8) * 512 + (t & 255)] = sG[(t >> 8) * 768 + 256 + (t & 255)];
  // ---- z' = att1(z) -> sG[:, :512]
  {
    const __bf16* Watt1 = (const __bf16*)(ws + OFF_WATT1);
    const float* att1b  = (const float*)(ws + OFF_ATT1B);
    for (int c = 0; c < 4; ++c) {
      mm_block<8>(sZ, 512, Watt1, 512, 16, c * 8, acc);
#pragma unroll
      for (int i = 0; i < 8; ++i) { int n = (c * 8 + i) * 16 + nl; float bb = att1b[n];
#pragma unroll
        for (int r = 0; r < 8; ++r) sG[(mbase + r) * 768 + n] = (__bf16)(acc[i][r] + bb); }
    }
  }
  // ---- fc1 512->64 (lrelu 0.01) -> sZ[:, :64]
  {
    const float* fc1b = (const float*)(ws + OFF_FC1B);
    mm_block<4>(sG, 768, (const __bf16*)(ws + OFF_WFC1), 64, 16, 0, acc);
#pragma unroll
    for (int i = 0; i < 4; ++i) { int n = i * 16 + nl; float bb = fc1b[n];
#pragma unroll
      for (int r = 0; r < 8; ++r) sZ[(mbase + r) * 512 + n] = (__bf16)lrelu_f(acc[i][r] + bb, 0.01f); }
  }
  // ---- fc2 64->16 (lrelu 0.01) -> sG[:, :16]; zero K-pad cols [16,32)
  {
    const float* fc2b = (const float*)(ws + OFF_FC2B);
    mm_block<1>(sZ, 512, (const __bf16*)(ws + OFF_WFC2), 16, 2, 0, acc);
    float bb = fc2b[nl];
#pragma unroll
    for (int r = 0; r < 8; ++r) sG[(mbase + r) * 768 + nl] = (__bf16)lrelu_f(acc[0][r] + bb, 0.01f);
    for (int t = lane; t < 256; t += 32) sG[(t >> 4) * 768 + 16 + (t & 15)] = (__bf16)0.f;
  }
  // ---- fc3 16->1 + sigmoid
  mm_block<1>(sG, 768, (const __bf16*)(ws + OFF_WFC3), 16, 1, 0, acc);
  if (nl == 0) {
    float b3 = *(const float*)(ws + OFF_FC3B);
#pragma unroll
    for (int r = 0; r < 8; ++r)
      out[r0 + mbase + r] = sigmoid_f(acc[0][r] + b3);
  }
}

// ---------------------------------------------------------------------------
// Prep kernels
// ---------------------------------------------------------------------------

// x -> bf16 [B][1728] with bn_in1 / bn_in2 folded per column; K padded w/ zeros.
__global__ void pack_x_kernel(const float* __restrict__ x, __bf16* __restrict__ xb, int B,
                              const float* g1, const float* b1, const float* m1, const float* v1,
                              const float* g2, const float* b2, const float* m2, const float* v2)
{
  size_t idx = (size_t)blockIdx.x * blockDim.x + threadIdx.x;
  size_t total = (size_t)B * 1728;
  if (idx >= total) return;
  int row = (int)(idx / 1728);
  int col = (int)(idx % 1728);
  float val = 0.f;
  if (col < 832) {
    if (col < 804) {
      float s = g1[col] * rsqrtf(v1[col] + 1e-5f);
      val = s * x[(size_t)row * 1672 + col] + (b1[col] - m1[col] * s);
    }
  } else {
    int j = col - 832;
    if (j < 868) {
      float s = g2[j] * rsqrtf(v2[j] + 1e-5f);
      val = s * x[(size_t)row * 1672 + 804 + j] + (b2[j] - m2[j] * s);
    }
  }
  xb[idx] = (__bf16)val;
}

// W (Cout,Cin,taps) center tap -> bf16 Wt[k][nOff+n] (row-major Kpad x Npad).
__global__ void pack_w_kernel(const float* __restrict__ src, __bf16* __restrict__ dst,
                              int Cin, int Cout, int taps, int tap, int Npad, int nOff, int Kpad)
{
  int idx = blockIdx.x * blockDim.x + threadIdx.x;
  int total = Kpad * Cout;
  if (idx >= total) return;
  int k = idx / Cout, n = idx % Cout;
  float v = (k < Cin) ? src[((size_t)n * Cin + k) * taps + tap] : 0.f;
  dst[(size_t)k * Npad + nOff + n] = (__bf16)v;
}

// Fold bn1/bn2 + conv biases into per-channel (scale, shift) pairs:
//   bn(acc + cb) = s*acc + (s*cb + t),  s = g*rsqrt(v+eps), t = b - m*s
__global__ void pack_epi_kernel(const float* c1_b, const float* c2_b,
                                const float* g1, const float* b1, const float* m1, const float* v1,
                                const float* g2, const float* b2, const float* m2, const float* v2,
                                float2* ep1, float2* ep2, float2* ep3, int C)
{
  int n = blockIdx.x * blockDim.x + threadIdx.x;
  if (n >= C) return;
  float s1 = g1[n] * rsqrtf(v1[n] + 1e-5f);
  float t1 = b1[n] - m1[n] * s1;
  float s2 = g2[n] * rsqrtf(v2[n] + 1e-5f);
  float t2 = b2[n] - m2[n] * s2;
  ep1[n] = make_float2(s1, s1 * c1_b[n] + t1);
  ep2[n] = make_float2(s1, s1 * c2_b[n] + t1);
  ep3[n] = make_float2(s2, s2 * c2_b[n] + t2);
}

// dst[0..n) = a, dst[n..2n) = b
__global__ void cat2_kernel(const float* a, const float* b, float* dst, int n)
{
  int i = blockIdx.x * blockDim.x + threadIdx.x;
  if (i < n) dst[i] = a[i];
  else if (i < 2 * n) dst[i] = b[i - n];
}

__global__ void copy_kernel(const float* a, float* dst, int n)
{
  int i = blockIdx.x * blockDim.x + threadIdx.x;
  if (i < n) dst[i] = a[i];
}

// ---------------------------------------------------------------------------
// Host launcher
// ---------------------------------------------------------------------------
extern "C" void kernel_launch(void* const* d_in, const int* in_sizes, int n_in,
                              void* d_out, int out_size, void* d_ws, size_t ws_size,
                              hipStream_t stream)
{
  const int B = out_size;  // 16384
  auto F = [&](int i) { return (const float*)d_in[i]; };
  char* ws = (char*)d_ws;

  // zero the packed-weight region once (covers all K/N padding)
  hipMemsetAsync((void*)ws, 0, OFF_WEND, stream);

  {
    size_t tot = (size_t)B * 1728;
    pack_x_kernel<<<dim3((unsigned)((tot + 255) / 256)), dim3(256), 0, stream>>>(
        F(0), (__bf16*)(ws + OFF_XB), B,
        F(1), F(2), F(3), F(4), F(5), F(6), F(7), F(8));
  }
  auto PW = [&](const float* src, size_t dstOff, int Cin, int Cout, int taps, int tap,
                int Npad, int nOff, int Kpad) {
    int tot = Kpad * Cout;
    pack_w_kernel<<<dim3((tot + 255) / 256), dim3(256), 0, stream>>>(
        src, (__bf16*)(ws + dstOff), Cin, Cout, taps, tap, Npad, nOff, Kpad);
  };
  // gru (57..62)
  PW(F(57), OFF_WG, 804, 384, 1, 0, 768, 0,   832);   // fw_wih
  PW(F(60), OFF_WG, 804, 384, 1, 0, 768, 384, 832);   // bw_wih
  // rb1 (c1_w=9 c1_b=10 c2_w=11 c2_b=12 rc1_w=13 rc1_b=14 rc2_w=15 rc2_b=16 bn1=17..20 bn2=21..24)
  PW(F(13), OFF_W1RC1, 868, 128, 1, 0, 128, 0, 896);
  PW(F(9),  OFF_W1C1,  868, 128, 7, 3, 128, 0, 896);
  PW(F(15), OFF_W1RC2, 128, 128, 7, 3, 128, 0, 128);
  PW(F(11), OFF_W1C2,  128, 128, 7, 3, 128, 0, 128);
  // rb2 (base 25)
  PW(F(29), OFF_W2RC1, 128, 256, 1, 0, 256, 0, 128);
  PW(F(25), OFF_W2C1,  128, 256, 7, 3, 256, 0, 128);
  PW(F(31), OFF_W2RC2, 256, 256, 7, 3, 256, 0, 256);
  PW(F(27), OFF_W2C2,  256, 256, 7, 3, 256, 0, 256);
  // rb3 (base 41; c1/c2 are k=5 -> tap 2, rc2 stays k=7 -> tap 3)
  PW(F(45), OFF_W3RC1, 256, 256, 1, 0, 256, 0, 256);
  PW(F(41), OFF_W3C1,  256, 256, 5, 2, 256, 0, 256);
  PW(F(47), OFF_W3RC2, 256, 256, 7, 3, 256, 0, 256);
  PW(F(43), OFF_W3C2,  256, 256, 5, 2, 256, 0, 256);
  // heads
  PW(F(63), OFF_WATT,  256, 256, 1, 0, 256, 0, 256);
  PW(F(65), OFF_WATT1, 512, 512, 1, 0, 512, 0, 512);
  PW(F(67), OFF_WFC1,  512, 64,  1, 0, 64,  0, 512);
  PW(F(69), OFF_WFC2,  64,  16,  1, 0, 16,  0, 64);
  PW(F(71), OFF_WFC3,  16,  1,   1, 0, 16,  0, 32);

  // folded epilogue constants
  pack_epi_kernel<<<dim3(1), dim3(128), 0, stream>>>(
      F(10), F(12), F(17), F(18), F(19), F(20), F(21), F(22), F(23), F(24),
      (float2*)(ws + OFF_EP1A), (float2*)(ws + OFF_EP2A), (float2*)(ws + OFF_EP3A), 128);
  pack_epi_kernel<<<dim3(1), dim3(256), 0, stream>>>(
      F(26), F(28), F(33), F(34), F(35), F(36), F(37), F(38), F(39), F(40),
      (float2*)(ws + OFF_EP1B), (float2*)(ws + OFF_EP2B), (float2*)(ws + OFF_EP3B), 256);
  pack_epi_kernel<<<dim3(1), dim3(256), 0, stream>>>(
      F(42), F(44), F(49), F(50), F(51), F(52), F(53), F(54), F(55), F(56),
      (float2*)(ws + OFF_EP1C), (float2*)(ws + OFF_EP2C), (float2*)(ws + OFF_EP3C), 256);
  cat2_kernel<<<dim3(3), dim3(256), 0, stream>>>(F(58), F(61), (float*)(ws + OFF_BG), 384);
  cat2_kernel<<<dim3(3), dim3(256), 0, stream>>>(F(59), F(62), (float*)(ws + OFF_BHH), 384);
  // raw biases into ws (fixed offsets -> kernel needs no d_in pointers)
  copy_kernel<<<dim3(1), dim3(128), 0, stream>>>(F(14), (float*)(ws + OFF_B1RC1), 128);
  copy_kernel<<<dim3(1), dim3(128), 0, stream>>>(F(16), (float*)(ws + OFF_B1RC2), 128);
  copy_kernel<<<dim3(1), dim3(256), 0, stream>>>(F(30), (float*)(ws + OFF_B2RC1), 256);
  copy_kernel<<<dim3(1), dim3(256), 0, stream>>>(F(32), (float*)(ws + OFF_B2RC2), 256);
  copy_kernel<<<dim3(1), dim3(256), 0, stream>>>(F(46), (float*)(ws + OFF_B3RC1), 256);
  copy_kernel<<<dim3(1), dim3(256), 0, stream>>>(F(48), (float*)(ws + OFF_B3RC2), 256);
  copy_kernel<<<dim3(1), dim3(256), 0, stream>>>(F(64), (float*)(ws + OFF_ATTB),  256);
  copy_kernel<<<dim3(2), dim3(256), 0, stream>>>(F(66), (float*)(ws + OFF_ATT1B), 512);
  copy_kernel<<<dim3(1), dim3(64),  0, stream>>>(F(68), (float*)(ws + OFF_FC1B),  64);
  copy_kernel<<<dim3(1), dim3(16),  0, stream>>>(F(70), (float*)(ws + OFF_FC2B),  16);
  copy_kernel<<<dim3(1), dim3(1),   0, stream>>>(F(72), (float*)(ws + OFF_FC3B),  1);

  fused_model<<<dim3(B / 16), dim3(32), 0, stream>>>(ws, (float*)d_out);
  (void)in_sizes; (void)n_in; (void)ws_size;
}
